// FrameSimilarity_33749853012753
// MI455X (gfx1250) — compile-verified, matrix operand order checked by reading, not run on hardware
//
#include <hip/hip_runtime.h>

// ---------------------------------------------------------------------------
// Problem constants (from the reference)
// ---------------------------------------------------------------------------
#define BB 8
#define CC 1024
#define TT 512
#define HW 49
#define DD 128      // sim_dim == out_dim == 128
#define WN 101      // lookup window
#define BT (BB * TT)        // 4096 rows total
#define NTILES (TT / 16)    // 32 row tiles per batch

// WMMA vector types (gfx1250, wave32)
typedef __attribute__((ext_vector_type(16))) __bf16 v16bf;
typedef __attribute__((ext_vector_type(8)))  float  v8f;

union FragU { unsigned int u[8]; v16bf v; };

__device__ __forceinline__ unsigned short f32_to_bf16(float f) {
    union { float f; unsigned int u; } cv; cv.f = f;
    unsigned int u = cv.u;
    u += 0x7FFFu + ((u >> 16) & 1u);   // round to nearest even
    return (unsigned short)(u >> 16);
}

// Async 16-byte global -> LDS copy (per lane), tracked by ASYNCcnt.
// lds_off is the byte offset from the workgroup's LDS base (our single
// __shared__ block per kernel starts at offset 0).
__device__ __forceinline__ void async_cp16(unsigned lds_off, const void* gptr) {
    asm volatile("global_load_async_to_lds_b128 %0, %1, off"
                 :: "v"(lds_off), "v"((unsigned long long)(size_t)gptr)
                 : "memory");
}
__device__ __forceinline__ void async_wait0() {
    asm volatile("s_wait_asynccnt 0x0" ::: "memory");
}

// ---------------------------------------------------------------------------
// Kernel 1: spatial mean over H*W=49 -> xbar[b*T+t][c] (bf16)
// Streams 822 MB (the whole problem's roofline). 50KB block staged with
// async b128 copies; stride-49 LDS reduction is bank-conflict-free.
// grid (T/256, C, B), block 256
// ---------------------------------------------------------------------------
__global__ void fs_mean_kernel(const float* __restrict__ in,
                               unsigned short* __restrict__ xbar) {
    __shared__ alignas(16) float lds[256 * HW];   // 50176 B = 3136 x 16B
    const int t0 = blockIdx.x * 256;
    const int c  = blockIdx.y;
    const int b  = blockIdx.z;
    const int tid = threadIdx.x;
    const float* base = in + ((size_t)(b * CC + c) * TT + t0) * HW;
    if (tid == 0) __builtin_prefetch(base + 256 * HW, 0, 1);  // global_prefetch_b8
#pragma unroll
    for (int k = 0; k < 12; ++k) {
        int i = tid + k * 256;
        async_cp16((unsigned)(i * 16), base + i * 4);
    }
    if (tid < 64) {
        int i = tid + 3072;                        // 3136 = 12*256 + 64
        async_cp16((unsigned)(i * 16), base + i * 4);
    }
    async_wait0();
    __syncthreads();
    float s = 0.0f;
#pragma unroll
    for (int j = 0; j < HW; ++j) s += lds[tid * HW + j];
    s *= (1.0f / 49.0f);
    xbar[(size_t)(b * TT + t0 + tid) * CC + c] = f32_to_bf16(s);
}

// ---------------------------------------------------------------------------
// Kernel 2a/2b: weight conversion to bf16, TRANSPOSED to [n][k] so that WMMA
// B-fragment gathers are contiguous uint loads from LDS. fc_w K-padded to 128.
// ---------------------------------------------------------------------------
__global__ void fs_cvt_projw(const float* __restrict__ w, unsigned short* __restrict__ o) {
    int i = blockIdx.x * 256 + threadIdx.x;        // i = n*CC + k
    if (i >= CC * DD) return;
    int n = i >> 10, k = i & (CC - 1);
    o[i] = f32_to_bf16(w[(size_t)k * DD + n]);
}
__global__ void fs_cvt_fcw(const float* __restrict__ w, unsigned short* __restrict__ o) {
    int i = blockIdx.x * 256 + threadIdx.x;        // i = n*DD + k
    if (i >= DD * DD) return;
    int n = i >> 7, k = i & 127;
    o[i] = (k < WN) ? f32_to_bf16(w[(size_t)k * DD + n]) : (unsigned short)0;
}

// ---------------------------------------------------------------------------
// Kernel 3: projection GEMM [4096x1024]@[1024x128] + bias, L2-normalize rows,
// store bf16 xn. 16 rows/block, 8 waves = 8 n-tiles, K in 32-chunks via
// async-to-LDS staging; v_wmma_f32_16x16x32_bf16 accumulate.
// ---------------------------------------------------------------------------
__global__ void fs_proj_kernel(const unsigned short* __restrict__ A,    // [4096][1024]
                               const unsigned short* __restrict__ Bt,   // [128][1024] (n-major)
                               const float* __restrict__ bias,
                               unsigned short* __restrict__ xn) {       // [4096][128]
    __shared__ alignas(16) char smem[1024 + 8192 + 8192 + 64];
    unsigned short* lA = (unsigned short*)smem;              // [16][32]   @0
    unsigned short* lB = (unsigned short*)(smem + 1024);     // [128][32]  @1024
    float* lO     = (float*)(smem + 1024 + 8192);            // [16][128]  @9216
    float* lScale = (float*)(smem + 1024 + 8192 + 8192);     // [16]       @17408

    const int row0 = blockIdx.x * 16;
    const int tid  = threadIdx.x;
    const int wid  = tid >> 5;
    const int lane = tid & 31;
    const int mrow = lane & 15;
    const int khi  = (lane >> 4) << 3;       // 0 or 8
    const int ncol = (wid << 4) + mrow;      // this wave's B column

    v8f acc = {};
    for (int k0 = 0; k0 < CC; k0 += 32) {
        // A tile: 16 rows x 64B = 64 chunks
        if (tid < 64)
            async_cp16((unsigned)(tid * 16),
                       A + (size_t)(row0 + (tid >> 2)) * CC + k0 + (tid & 3) * 8);
        // B tile (transposed): 128 n x 64B = 512 chunks
#pragma unroll
        for (int it = 0; it < 2; ++it) {
            int i = tid + it * 256;
            async_cp16((unsigned)(1024 + i * 16),
                       Bt + (size_t)(i >> 2) * CC + k0 + (i & 3) * 8);
        }
        async_wait0();
        __syncthreads();
        FragU a, bfr;
#pragma unroll
        for (int v = 0; v < 8; ++v) {
            int kk = ((v < 4) ? (2 * v) : (16 + 2 * (v - 4))) + khi;
            a.u[v]   = *(const unsigned int*)&lA[mrow * 32 + kk];
            bfr.u[v] = *(const unsigned int*)&lB[ncol * 32 + kk];
        }
        acc = __builtin_amdgcn_wmma_f32_16x16x32_bf16(false, a.v, false, bfr.v,
                                                      (short)0, acc, false, false);
        __syncthreads();
    }
    // D layout: VGPR i, lanes 0-15 -> M=i; lanes 16-31 -> M=8+i; N = lane&15
    {
        const int nc = (wid << 4) + (lane & 15);
        const int mbase = (lane >> 4) << 3;
        const float bv = bias[nc];
#pragma unroll
        for (int i = 0; i < 8; ++i) lO[(i + mbase) * DD + nc] = acc[i] + bv;
    }
    __syncthreads();
    if (tid < 16) {
        float s = 0.0f;
        for (int j = 0; j < DD; ++j) { float v = lO[tid * DD + j]; s += v * v; }
        lScale[tid] = 1.0f / fmaxf(sqrtf(s), 1e-12f);
    }
    __syncthreads();
    for (int i = tid; i < 16 * DD; i += 256) {
        int m = i >> 7;
        xn[(size_t)(row0 + m) * DD + (i & 127)] = f32_to_bf16(lO[i] * lScale[m]);
    }
}

// ---------------------------------------------------------------------------
// Kernel 4: banded similarity strips: strip[b][tile][16][128] =
// xn[T0..T0+15] . xn[T0-56+j]^T  (out-of-range rows zero == reference pad).
// In-range rows async-copied; out-of-range rows zero-filled with b128 stores.
// grid (32, 8), block 256
// ---------------------------------------------------------------------------
__global__ void fs_sim_kernel(const unsigned short* __restrict__ xn,
                              float* __restrict__ strip) {
    __shared__ alignas(16) char smem[4096 + 32768];
    unsigned short* lA = (unsigned short*)smem;            // [16][128]  @0
    unsigned short* lB = (unsigned short*)(smem + 4096);   // [128][128] @4096
    const int tile = blockIdx.x;
    const int b    = blockIdx.y;
    const int T0   = tile * 16;
    const int tid  = threadIdx.x;
    const int wid  = tid >> 5;
    const int lane = tid & 31;
    const unsigned short* xb = xn + (size_t)b * TT * DD;

    // lA: 16 rows x 256B = 256 chunks (one per thread)
    async_cp16((unsigned)(tid * 16),
               xb + (size_t)(T0 + (tid >> 4)) * DD + (tid & 15) * 8);
    // lB: 128 rows x 256B = 2048 chunks; zero-fill out-of-range rows
#pragma unroll
    for (int it = 0; it < 8; ++it) {
        int i = tid + it * 256;
        int s = T0 - 56 + (i >> 4);
        if (s >= 0 && s < TT) {
            async_cp16((unsigned)(4096 + i * 16),
                       xb + (size_t)s * DD + (i & 15) * 8);
        } else {
            uint4 z = {0u, 0u, 0u, 0u};
            *(uint4*)(smem + 4096 + i * 16) = z;           // ds_store_b128
        }
    }
    async_wait0();
    __syncthreads();

    v8f acc = {};
    const int mrow = lane & 15;
    const int khi  = (lane >> 4) << 3;
    const int nrow = (wid << 4) + mrow;    // B column n == x row (s0+n): K-contiguous
#pragma unroll
    for (int kc = 0; kc < 4; ++kc) {
        FragU a, bfr;
#pragma unroll
        for (int v = 0; v < 8; ++v) {
            int kk = kc * 32 + ((v < 4) ? (2 * v) : (16 + 2 * (v - 4))) + khi;
            a.u[v]   = *(const unsigned int*)&lA[mrow * DD + kk];
            bfr.u[v] = *(const unsigned int*)&lB[nrow * DD + kk];
        }
        acc = __builtin_amdgcn_wmma_f32_16x16x32_bf16(false, a.v, false, bfr.v,
                                                      (short)0, acc, false, false);
    }
    const int nc = (wid << 4) + (lane & 15);
    const int mbase = (lane >> 4) << 3;
    float* dst = strip + (size_t)(b * NTILES + tile) * 16 * DD;
#pragma unroll
    for (int i = 0; i < 8; ++i) dst[(size_t)(i + mbase) * DD + nc] = acc[i];
}

// ---------------------------------------------------------------------------
// Kernel 5: band[t][w] = strip[b][t/16][t%16][(t%16)+w+6] for w<101, 0-padded
// to 128; bf16 FC A-operand. grid 4096, block 128.
// ---------------------------------------------------------------------------
__global__ void fs_band_kernel(const float* __restrict__ strip,
                               unsigned short* __restrict__ band) {
    const int t = blockIdx.x;              // b*512 + tt
    const int w = threadIdx.x;             // 0..127
    const int b = t >> 9, tt = t & 511;
    const int tile = tt >> 4, r = tt & 15;
    float v = 0.0f;
    if (w < WN)
        v = strip[((size_t)(b * NTILES + tile) * 16 + r) * DD + (r + w + 6)];
    band[(size_t)t * DD + w] = f32_to_bf16(v);
}

// ---------------------------------------------------------------------------
// Kernel 6: FC [4096x128]@[128x128] + bias, ReLU -> f32 out.
// fcw_t is [n][k] so both fragment gathers are contiguous uint LDS loads.
// ---------------------------------------------------------------------------
__global__ void fs_fc_kernel(const unsigned short* __restrict__ band,
                             const unsigned short* __restrict__ fcwt,  // [128][128] n-major
                             const float* __restrict__ bias,
                             float* __restrict__ out) {
    __shared__ alignas(16) char smem[4096 + 32768];
    unsigned short* lA = (unsigned short*)smem;            // [16][128]  @0
    unsigned short* lB = (unsigned short*)(smem + 4096);   // [128][128] @4096
    const int row0 = blockIdx.x * 16;
    const int tid  = threadIdx.x;
    const int wid  = tid >> 5;
    const int lane = tid & 31;

    async_cp16((unsigned)(tid * 16),
               band + (size_t)(row0 + (tid >> 4)) * DD + (tid & 15) * 8);
#pragma unroll
    for (int it = 0; it < 8; ++it) {
        int i = tid + it * 256;
        async_cp16((unsigned)(4096 + i * 16), fcwt + i * 8);
    }
    async_wait0();
    __syncthreads();

    v8f acc = {};
    const int mrow = lane & 15;
    const int khi  = (lane >> 4) << 3;
    const int ncol = (wid << 4) + mrow;
#pragma unroll
    for (int kc = 0; kc < 4; ++kc) {
        FragU a, bfr;
#pragma unroll
        for (int v = 0; v < 8; ++v) {
            int kk = kc * 32 + ((v < 4) ? (2 * v) : (16 + 2 * (v - 4))) + khi;
            a.u[v]   = *(const unsigned int*)&lA[mrow * DD + kk];
            bfr.u[v] = *(const unsigned int*)&lB[ncol * DD + kk];
        }
        acc = __builtin_amdgcn_wmma_f32_16x16x32_bf16(false, a.v, false, bfr.v,
                                                      (short)0, acc, false, false);
    }
    const int nc = (wid << 4) + (lane & 15);
    const int mbase = (lane >> 4) << 3;
    const float bv = bias[nc];
#pragma unroll
    for (int i = 0; i < 8; ++i) {
        float v = acc[i] + bv;
        out[(size_t)(row0 + i + mbase) * DD + nc] = fmaxf(v, 0.0f);
    }
}

// ---------------------------------------------------------------------------
// Workspace layout (bytes):
//   xbar   bf16 [4096][1024]     @ 0         (8388608)
//   projwt bf16 [128][1024]      @ 8388608   (262144)   (n-major)
//   fcwt   bf16 [128][128]       @ 8650752   (32768)    (n-major, K-padded)
//   xn     bf16 [4096][128]      @ 8683520   (1048576)
//   strip  f32  [8][32][16][128] @ 9732096   (2097152)
//   band   bf16 [4096][128]      @ 11829248  (1048576)
// total ~12.3 MB
// ---------------------------------------------------------------------------
extern "C" void kernel_launch(void* const* d_in, const int* in_sizes, int n_in,
                              void* d_out, int out_size, void* d_ws, size_t ws_size,
                              hipStream_t stream) {
    (void)in_sizes; (void)n_in; (void)out_size; (void)ws_size;
    const float* inputs = (const float*)d_in[0];
    const float* proj_w = (const float*)d_in[1];
    const float* proj_b = (const float*)d_in[2];
    const float* fc_w   = (const float*)d_in[3];
    const float* fc_b   = (const float*)d_in[4];
    float* out = (float*)d_out;

    char* ws = (char*)d_ws;
    unsigned short* xbar   = (unsigned short*)(ws + 0);
    unsigned short* projwt = (unsigned short*)(ws + 8388608);
    unsigned short* fcwt   = (unsigned short*)(ws + 8650752);
    unsigned short* xn     = (unsigned short*)(ws + 8683520);
    float*          strip  = (float*)        (ws + 9732096);
    unsigned short* band   = (unsigned short*)(ws + 11829248);

    fs_mean_kernel<<<dim3(TT / 256, CC, BB), 256, 0, stream>>>(inputs, xbar);
    fs_cvt_projw<<<(CC * DD + 255) / 256, 256, 0, stream>>>(proj_w, projwt);
    fs_cvt_fcw<<<(DD * DD + 255) / 256, 256, 0, stream>>>(fc_w, fcwt);
    fs_proj_kernel<<<BT / 16, 256, 0, stream>>>(xbar, projwt, proj_b, xn);
    fs_sim_kernel<<<dim3(NTILES, BB), 256, 0, stream>>>(xn, strip);
    fs_band_kernel<<<BT, 128, 0, stream>>>(strip, band);
    fs_fc_kernel<<<BT / 16, 256, 0, stream>>>(band, fcwt, fc_b, out);
}
